// GAT_35914516529780
// MI455X (gfx1250) — matrix operand order repaired and checked by reading, not compiled, not run on hardware
//
#include <hip/hip_runtime.h>
#include <hip/hip_bf16.h>

#define NEG_SLOPE 0.2f
#define GAT_EPS   1e-16f

typedef __attribute__((ext_vector_type(16))) __bf16 v16bf;
typedef __attribute__((ext_vector_type(8)))  __bf16 v8bf;
typedef __attribute__((ext_vector_type(8)))  float  v8f;

// ---------------- helpers: ordered-uint encoding for float atomicMax ----------------
__device__ __forceinline__ unsigned f2ord(float f) {
  unsigned u = __float_as_uint(f);
  return (u >> 31) ? ~u : (u | 0x80000000u);
}
__device__ __forceinline__ float ord2f(unsigned u) {
  return __uint_as_float((u >> 31) ? (u ^ 0x80000000u) : ~u);
}

// ---------------- elementwise utility kernels ----------------
__global__ void f32_to_bf16_kernel(const float* __restrict__ in, __bf16* __restrict__ out, int n) {
  int i = blockIdx.x * blockDim.x + threadIdx.x;
  if (i < n) out[i] = (__bf16)in[i];
}

__global__ void zero_f32_kernel(float* __restrict__ p, int n) {
  int i = blockIdx.x * blockDim.x + threadIdx.x;
  if (i < n) p[i] = 0.f;
}

// Pack row-major f32 weight [K, Ncols] into bf16 WMMA B-fragment layout:
// fragment index = ((kt*ntiles + nt)*32 + lane)*16 + j
// lane -> column (lane&15), lane>>4 -> K half; j -> K = kt*32 + (lane>>4)*8 + (j&7) + (j>>3)*16
__global__ void pack_w_kernel(const float* __restrict__ W, __bf16* __restrict__ out,
                              int K, int Ncols) {
  int idx = blockIdx.x * blockDim.x + threadIdx.x;
  int ktiles = K >> 5, ntiles = Ncols >> 4;
  int total = ktiles * ntiles * 32;
  if (idx >= total) return;
  int lane = idx & 31, tile = idx >> 5;
  int nt = tile % ntiles, kt = tile / ntiles;
  int n = (nt << 4) + (lane & 15);
  int kbase = (kt << 5) + ((lane >> 4) << 3);
  __bf16* o = out + (size_t)idx * 16;
#pragma unroll
  for (int j = 0; j < 16; ++j)
    o[j] = (__bf16)W[(size_t)(kbase + (j & 7) + ((j >> 3) << 4)) * Ncols + n];
}

// ---------------- WMMA GEMM: C[M,Ncols] = A_bf16[M,K] @ Bpacked (+bias) ----------------
// One wave computes a 16x64 strip (4 n-tiles share the A fragment). K, Ncols multiples of 32/64.
__global__ void gemm_bf16_wmma(const __bf16* __restrict__ A,
                               const __bf16* __restrict__ Bp,
                               const float* __restrict__ bias,
                               float* __restrict__ C,
                               int M, int K, int Ncols) {
  const int lane = threadIdx.x & 31;
  const int wave = blockIdx.x * (blockDim.x >> 5) + (threadIdx.x >> 5);
  const int ntiles = Ncols >> 4;
  const int ngroups = ntiles >> 2;            // groups of 4 n-tiles (64 cols)
  const int mtiles = (M + 15) >> 4;
  if (wave >= mtiles * ngroups) return;       // wave-uniform: EXEC stays all-1s for WMMA
  const int mt = wave / ngroups;
  const int ng = wave % ngroups;

  v8f acc[4] = {v8f{}, v8f{}, v8f{}, v8f{}};

  int m = (mt << 4) + (lane & 15);
  if (m > M - 1) m = M - 1;                   // clamp loads (stores guarded below)
  const int kb0 = (lane >> 4) << 3;           // A layout: two 8-elem runs per lane
  const __bf16* Arow = A + (size_t)m * K;
  const int ksteps = K >> 5;

  for (int kt = 0; kt < ksteps; ++kt) {
    const int kb = (kt << 5) + kb0;
    v8bf alo = *(const v8bf*)(Arow + kb);
    v8bf ahi = *(const v8bf*)(Arow + kb + 16);
    v16bf a;
#pragma unroll
    for (int j = 0; j < 8; ++j) { a[j] = alo[j]; a[8 + j] = ahi[j]; }
#pragma unroll
    for (int t = 0; t < 4; ++t) {
      const int nt = (ng << 2) + t;
      v16bf b = *(const v16bf*)(Bp + (((size_t)kt * ntiles + nt) * 32 + lane) * 16);
      acc[t] = __builtin_amdgcn_wmma_f32_16x16x32_bf16(
          /*neg_a=*/false, a, /*neg_b=*/false, b,
          /*c_mod=*/(short)0, acc[t], /*reuse_a=*/false, /*reuse_b=*/false);
    }
  }

  // C/D layout: VGPR r -> row = mt*16 + (lane>>4)*8 + r ; col = tile*16 + (lane&15)
  const int mrow = (mt << 4) + ((lane >> 4) << 3);
  const int ncol0 = (ng << 6) + (lane & 15);
#pragma unroll
  for (int t = 0; t < 4; ++t) {
    const int n = ncol0 + (t << 4);
    const float bv = bias ? bias[n] : 0.f;
#pragma unroll
    for (int r = 0; r < 8; ++r) {
      const int row = mrow + r;
      if (row < M) C[(size_t)row * Ncols + n] = acc[t][r] + bv;
    }
  }
}

// ---------------- attention coefficients per node/head ----------------
// h: [N, 2*128] (heads concatenated). alS/alD: [N,2]
__global__ void alpha_dots_kernel(const float* __restrict__ h,
                                  const float* __restrict__ a_src,
                                  const float* __restrict__ a_dst,
                                  float* __restrict__ alS, float* __restrict__ alD, int N) {
  int idx = blockIdx.x * blockDim.x + threadIdx.x;
  if (idx >= N * 2) return;
  int n = idx >> 1, hd = idx & 1;
  const float4* hp = (const float4*)(h + (size_t)n * 256 + hd * 128);
  const float4* sp = (const float4*)(a_src + hd * 128);
  const float4* dp = (const float4*)(a_dst + hd * 128);
  float s = 0.f, d = 0.f;
#pragma unroll 8
  for (int j = 0; j < 32; ++j) {
    float4 hv = hp[j], sv = sp[j], dv = dp[j];
    s += hv.x * sv.x + hv.y * sv.y + hv.z * sv.z + hv.w * sv.w;
    d += hv.x * dv.x + hv.y * dv.y + hv.z * dv.z + hv.w * dv.w;
  }
  alS[idx] = s; alD[idx] = d;
}

__global__ void init_stats_kernel(unsigned* __restrict__ emax, float* __restrict__ denom, int n2) {
  int i = blockIdx.x * blockDim.x + threadIdx.x;
  if (i < n2) { emax[i] = 0x007FFFFFu /* f2ord(-inf) */; denom[i] = 0.f; }
}

// pass 1: e = leaky_relu(alpha_s[src] + alpha_d[dst]); segment max over dst
__global__ void edge_lrelu_max_kernel(const int* __restrict__ ei, int E, int N,
                                      const float* __restrict__ alS,
                                      const float* __restrict__ alD,
                                      float* __restrict__ eedge,
                                      unsigned* __restrict__ emax) {
  int i = blockIdx.x * blockDim.x + threadIdx.x;
  int Etot = E + N;
  if (i >= Etot) return;
  if (i + 4096 < E) {                       // stream the edge list -> global_prefetch_b8
    __builtin_prefetch(ei + i + 4096, 0, 0);
    __builtin_prefetch(ei + E + i + 4096, 0, 0);
  }
  int s, d;
  if (i < E) { s = ei[i]; d = ei[E + i]; } else { s = d = i - E; }
#pragma unroll
  for (int hd = 0; hd < 2; ++hd) {
    float e = alS[s * 2 + hd] + alD[d * 2 + hd];
    e = (e > 0.f) ? e : NEG_SLOPE * e;
    eedge[(size_t)i * 2 + hd] = e;
    atomicMax(&emax[d * 2 + hd], f2ord(e));
  }
}

// decode ordered-uint max -> float in place; non-finite -> 0 (matches reference)
__global__ void decode_max_kernel(unsigned* __restrict__ emax, int n2) {
  int i = blockIdx.x * blockDim.x + threadIdx.x;
  if (i >= n2) return;
  float f = ord2f(emax[i]);
  if (!isfinite(f)) f = 0.f;
  ((float*)emax)[i] = f;
}

// pass 2: w = exp(e - emax[dst]); segment sum over dst; store w back over e
__global__ void edge_exp_sum_kernel(const int* __restrict__ ei, int E, int N,
                                    float* __restrict__ eedge,
                                    const float* __restrict__ emaxf,
                                    float* __restrict__ denom) {
  int i = blockIdx.x * blockDim.x + threadIdx.x;
  int Etot = E + N;
  if (i >= Etot) return;
  int d = (i < E) ? ei[E + i] : (i - E);
#pragma unroll
  for (int hd = 0; hd < 2; ++hd) {
    float w = __expf(eedge[(size_t)i * 2 + hd] - emaxf[d * 2 + hd]);
    eedge[(size_t)i * 2 + hd] = w;
    atomicAdd(&denom[d * 2 + hd], w);
  }
}

// pass 3: out[dst] += h[src] * (w / (denom[dst]+eps)). One wave per edge,
// 8 channels per lane (lanes 0-15: head 0, lanes 16-31: head 1).
__global__ void edge_aggregate_kernel(const int* __restrict__ ei, int E, int N,
                                      const float* __restrict__ h,
                                      const float* __restrict__ eedge,
                                      const float* __restrict__ denom,
                                      float* __restrict__ hacc) {
  int wave = blockIdx.x * (blockDim.x >> 5) + (threadIdx.x >> 5);
  int lane = threadIdx.x & 31;
  int Etot = E + N;
  if (wave >= Etot) return;
  int s, d;
  if (wave < E) { s = ei[wave]; d = ei[E + wave]; } else { s = d = wave - E; }
  int hd = lane >> 4;
  float alpha = eedge[(size_t)wave * 2 + hd] / (denom[d * 2 + hd] + GAT_EPS);
  int c0 = lane * 8;
  const float4* hp = (const float4*)(h + (size_t)s * 256 + c0);
  float4 v0 = hp[0], v1 = hp[1];
  float* o = hacc + (size_t)d * 256 + c0;
  atomicAdd(o + 0, v0.x * alpha);
  atomicAdd(o + 1, v0.y * alpha);
  atomicAdd(o + 2, v0.z * alpha);
  atomicAdd(o + 3, v0.w * alpha);
  atomicAdd(o + 4, v1.x * alpha);
  atomicAdd(o + 5, v1.y * alpha);
  atomicAdd(o + 6, v1.z * alpha);
  atomicAdd(o + 7, v1.w * alpha);
}

// epilogue: +bias, relu, emit bf16 for the next WMMA GEMM
__global__ void bias_relu_bf16_kernel(const float* __restrict__ acc,
                                      const float* __restrict__ bias,
                                      __bf16* __restrict__ out, int n, int cols) {
  int i = blockIdx.x * blockDim.x + threadIdx.x;
  if (i >= n) return;
  float v = acc[i] + bias[i % cols];
  out[i] = (__bf16)(v > 0.f ? v : 0.f);
}

// final head: sigmoid(h3[n,:] . Wp2 + bp2)
__global__ void head_out_kernel(const float* __restrict__ h3, const float* __restrict__ Wp2,
                                const float* __restrict__ bp2, float* __restrict__ out, int N) {
  int n = blockIdx.x * blockDim.x + threadIdx.x;
  if (n >= N) return;
  const float4* hp = (const float4*)(h3 + (size_t)n * 128);
  const float4* wp = (const float4*)Wp2;
  float t = bp2[0];
#pragma unroll 8
  for (int j = 0; j < 32; ++j) {
    float4 a = hp[j], b = wp[j];
    t += a.x * b.x + a.y * b.y + a.z * b.z + a.w * b.w;
  }
  out[n] = 1.f / (1.f + __expf(-t));
}

// ---------------- host launcher ----------------
extern "C" void kernel_launch(void* const* d_in, const int* in_sizes, int n_in,
                              void* d_out, int out_size, void* d_ws, size_t ws_size,
                              hipStream_t stream) {
  const float* x   = (const float*)d_in[0];
  const int*   ei  = (const int*)d_in[1];
  const float* W1  = (const float*)d_in[2];
  const float* as1 = (const float*)d_in[3];
  const float* ad1 = (const float*)d_in[4];
  const float* b1  = (const float*)d_in[5];
  const float* W2  = (const float*)d_in[6];
  const float* as2 = (const float*)d_in[7];
  const float* ad2 = (const float*)d_in[8];
  const float* b2  = (const float*)d_in[9];
  const float* Wp1 = (const float*)d_in[10];
  const float* bp1 = (const float*)d_in[11];
  const float* Wp2 = (const float*)d_in[12];
  const float* bp2 = (const float*)d_in[13];

  const int IN = 128, HH = 256, HID = 128;
  const int N = in_sizes[0] / IN;
  const int E = in_sizes[1] / 2;
  const int Etot = E + N;

  char* ws = (char*)d_ws;
  size_t off = 0;
  auto alloc = [&](size_t bytes) -> void* {
    void* p = ws + off;
    off = (off + bytes + 255) & ~(size_t)255;
    return p;
  };
  __bf16*  xbf   = (__bf16*) alloc((size_t)N * IN * 2);
  float*   hbuf  = (float*)  alloc((size_t)N * HH * 4);
  float*   hacc  = (float*)  alloc((size_t)N * HH * 4);
  __bf16*  hbf   = (__bf16*) alloc((size_t)N * HH * 2);
  float*   h3    = (float*)  alloc((size_t)N * HID * 4);
  float*   alS   = (float*)  alloc((size_t)N * 2 * 4);
  float*   alD   = (float*)  alloc((size_t)N * 2 * 4);
  unsigned* emaxU= (unsigned*)alloc((size_t)N * 2 * 4);
  float*   denom = (float*)  alloc((size_t)N * 2 * 4);
  float*   eedge = (float*)  alloc((size_t)Etot * 2 * 4);
  __bf16*  W1p   = (__bf16*) alloc((size_t)IN * HH * 2);
  __bf16*  W2p   = (__bf16*) alloc((size_t)HH * HH * 2);
  __bf16*  Wp1p  = (__bf16*) alloc((size_t)HH * HID * 2);
  (void)ws_size; (void)n_in; (void)out_size;

  const int B = 256;
  auto cdiv = [](int a, int b) { return (a + b - 1) / b; };

  // one-time conversions / weight packing (re-done every call: deterministic)
  f32_to_bf16_kernel<<<cdiv(N * IN, B), B, 0, stream>>>(x, xbf, N * IN);
  pack_w_kernel<<<cdiv((IN / 32) * (HH / 16) * 32, B), B, 0, stream>>>(W1, W1p, IN, HH);
  pack_w_kernel<<<cdiv((HH / 32) * (HH / 16) * 32, B), B, 0, stream>>>(W2, W2p, HH, HH);
  pack_w_kernel<<<cdiv((HH / 32) * (HID / 16) * 32, B), B, 0, stream>>>(Wp1, Wp1p, HH, HID);

  const __bf16* gin = xbf;
  int gk = IN;
  const __bf16* Wpk[2] = {W1p, W2p};
  const float*  aS[2]  = {as1, as2};
  const float*  aD[2]  = {ad1, ad2};
  const float*  bb[2]  = {b1, b2};

  for (int layer = 0; layer < 2; ++layer) {
    // h = x @ W  (pre-attention features, f32 accumulate)
    int waves = ((N + 15) / 16) * (HH / 64);
    gemm_bf16_wmma<<<cdiv(waves * 32, B), B, 0, stream>>>(gin, Wpk[layer], nullptr, hbuf, N, gk, HH);
    // per-node attention dots
    alpha_dots_kernel<<<cdiv(N * 2, B), B, 0, stream>>>(hbuf, aS[layer], aD[layer], alS, alD, N);
    // reset stats + accumulator every call
    init_stats_kernel<<<cdiv(N * 2, B), B, 0, stream>>>(emaxU, denom, N * 2);
    zero_f32_kernel<<<cdiv(N * HH, B), B, 0, stream>>>(hacc, N * HH);
    // segment softmax over destination
    edge_lrelu_max_kernel<<<cdiv(Etot, B), B, 0, stream>>>(ei, E, N, alS, alD, eedge, emaxU);
    decode_max_kernel<<<cdiv(N * 2, B), B, 0, stream>>>(emaxU, N * 2);
    edge_exp_sum_kernel<<<cdiv(Etot, B), B, 0, stream>>>(ei, E, N, eedge, (const float*)emaxU, denom);
    // weighted scatter-add (wave per edge)
    edge_aggregate_kernel<<<cdiv(Etot, 8), B, 0, stream>>>(ei, E, N, hbuf, eedge, denom, hacc);
    // +bias, relu, down-convert for next GEMM
    bias_relu_bf16_kernel<<<cdiv(N * HH, B), B, 0, stream>>>(hacc, bb[layer], hbf, N * HH, HH);
    gin = hbf;
    gk = HH;
  }

  // post_mp: Linear(256,128)+bias, then Linear(128,1)+sigmoid
  int waves = ((N + 15) / 16) * (HID / 64);
  gemm_bf16_wmma<<<cdiv(waves * 32, B), B, 0, stream>>>(hbf, Wp1p, bp1, h3, N, HH, HID);
  head_out_kernel<<<cdiv(N, B), B, 0, stream>>>(h3, Wp2, bp2, (float*)d_out, N);
}
